// HyperbolicPatchEmbedding_5952824672956
// MI455X (gfx1250) — matrix operand order repaired and checked by reading, not compiled, not run on hardware
//
#include <hip/hip_runtime.h>

// ---------------------------------------------------------------------------
// HyperbolicPatchEmbedding for MI455X (gfx1250, wave32, WMMA).
//
// Pipeline:
//   1) patchify: x[64,3,224,224] f32 -> patches[12544,768] bf16 (+ row norms)
//   2) convertW: W[768,768] f32 -> bf16 (B operand of patches @ W.T is just W)
//   3) reduceB : b2 = sum(b^2)
//   4) gemm    : mx = patches @ W.T via v_wmma_f32_16x16x32_bf16,
//                double-buffered LDS fed by GLOBAL_LOAD_ASYNC_TO_LDS_B128
//                (ASYNCcnt-tracked), overlapping copy of tile k+1 with WMMA
//                on tile k.
//   5) epilogue: per-row Mobius matvec scale + Mobius add + project, fused as
//                out = s_m * mx + s_b * b  (scalars from row reductions)
// ---------------------------------------------------------------------------

typedef __attribute__((ext_vector_type(16))) __bf16 v16bf;
typedef __attribute__((ext_vector_type(8)))  float  v8f;
typedef __attribute__((ext_vector_type(4)))  int    v4i;

#define NROWS   12544      // 64 * 196 patch rows
#define KDIM    768
#define NDIM    768
#define KSTEPS  (KDIM / 32)
#define MAXNORM 0.996f     // (1 - 4e-3) / sqrt(c), c = 1
#define EPSN    1e-15f

// ---- CDNA5 async global->LDS copy path (ASYNCcnt) -------------------------
#if defined(__gfx1250__) && __has_builtin(__builtin_amdgcn_global_load_async_to_lds_b128)
#define USE_ASYNC_LDS 1
typedef __attribute__((address_space(1))) v4i* gv4i_p;   // global v4i*
typedef __attribute__((address_space(3))) v4i* lv4i_p;   // LDS v4i*
__device__ __forceinline__ void async_g2l_b128(const void* g, void* l) {
  __builtin_amdgcn_global_load_async_to_lds_b128(
      (gv4i_p)g, (lv4i_p)l, /*offset=*/0, /*cpol=*/0);
}
#if __has_builtin(__builtin_amdgcn_s_wait_asynccnt)
#define WAIT_ASYNC(n) __builtin_amdgcn_s_wait_asynccnt(n)
#else
#define WAIT_ASYNC(n) asm volatile("s_wait_asynccnt %0" ::"n"(n) : "memory")
#endif
#else
#define USE_ASYNC_LDS 0
#endif

// ---------------- wave / block reduction helper ----------------
__device__ inline float waveReduceSum(float v) {
  #pragma unroll
  for (int off = 16; off > 0; off >>= 1) v += __shfl_down(v, off, 32);
  return v;
}

// ---------------- 1) patchify + row norm ----------------
__global__ __launch_bounds__(256)
void patchify_kernel(const float* __restrict__ x,
                     __bf16* __restrict__ patches,
                     float* __restrict__ xnorm) {
  const int r = blockIdx.x;              // patch row 0..12543
  const int t = threadIdx.x;
  const int bimg = r / 196;
  const int pi   = r % 196;
  const int hp   = pi / 14, wp = pi % 14;

  float ss = 0.f;
  #pragma unroll
  for (int j = 0; j < 3; ++j) {
    const int d   = t + j * 256;         // 0..767 within patch row
    const int c   = d >> 8;              // channel
    const int rem = d & 255;
    const int py  = rem >> 4, px = rem & 15;
    const size_t src = ((size_t)(bimg * 3 + c) * 224 + (size_t)(hp * 16 + py)) * 224
                     + (size_t)(wp * 16 + px);
    const float v = x[src];
    patches[(size_t)r * KDIM + d] = (__bf16)v;
    ss += v * v;
  }
  ss = waveReduceSum(ss);
  __shared__ float red[8];
  const int w = t >> 5, l = t & 31;
  if (l == 0) red[w] = ss;
  __syncthreads();
  if (t == 0) {
    float tot = 0.f;
    #pragma unroll
    for (int i = 0; i < 8; ++i) tot += red[i];
    xnorm[r] = fmaxf(sqrtf(tot), EPSN);
  }
}

// ---------------- 2) W f32 -> bf16 ----------------
__global__ __launch_bounds__(256)
void convertW_kernel(const float* __restrict__ W, __bf16* __restrict__ Wbf) {
  const int i = blockIdx.x * 256 + threadIdx.x;   // < 589824
  Wbf[i] = (__bf16)W[i];
}

// ---------------- 3) b2 = sum(b^2) ----------------
__global__ __launch_bounds__(256)
void reduceB_kernel(const float* __restrict__ b, float* __restrict__ b2) {
  const int t = threadIdx.x;
  float ss = 0.f;
  #pragma unroll
  for (int j = 0; j < 3; ++j) { float v = b[t + j * 256]; ss += v * v; }
  ss = waveReduceSum(ss);
  __shared__ float red[8];
  const int w = t >> 5, l = t & 31;
  if (l == 0) red[w] = ss;
  __syncthreads();
  if (t == 0) {
    float tot = 0.f;
    #pragma unroll
    for (int i = 0; i < 8; ++i) tot += red[i];
    *b2 = tot;
  }
}

// ---------------- 4) WMMA GEMM: mx[m][n] = sum_k patches[m][k] * W[n][k] ----
// Block tile 128x128, 8 waves laid out 4x2, wave tile 32x64 (2x4 frags).
// LDS tiles (x2 buffers): sA[128 rows][32 k] bf16, sB[128 n][32 k] bf16.
__global__ __launch_bounds__(256)
void gemm_kernel(const __bf16* __restrict__ A,    // [12544][768]
                 const __bf16* __restrict__ Bw,   // [768][768] (= bf16(W))
                 float* __restrict__ C) {         // [12544][768]
  __shared__ ushort sA[2][128 * 32];
  __shared__ ushort sB[2][128 * 32];

  const int tid  = threadIdx.x;
  const int bm   = blockIdx.y;          // 0..97
  const int bn   = blockIdx.x;          // 0..5
  const int wave = tid >> 5;
  const int lane = tid & 31;
  const int wm   = wave >> 1;           // 0..3 -> M offset wm*32
  const int wn   = wave & 1;            // 0..1 -> N offset wn*64
  const int lr   = lane & 15;
  const int lh   = lane >> 4;

  v8f acc[2][4] = {};                   // zero-initialized accumulators

  // cooperative tile-load mapping: each thread moves 32B (16 bf16) per tile
  const int row = tid >> 1;             // 0..127
  const int seg = tid & 1;              // 0/1 -> k sub-segment of 16
  const uint4* gA = (const uint4*)(A + ((size_t)(bm * 128 + row)) * KDIM + seg * 16);
  const uint4* gB = (const uint4*)(Bw + ((size_t)(bn * 128 + row)) * KDIM + seg * 16);
  const int sOff = row * 32 + seg * 16;

#if USE_ASYNC_LDS
  // ---- prologue: async-copy tile 0 into buffer 0 (4 ops / thread / tile)
  async_g2l_b128(&gA[0], &sA[0][sOff]);
  async_g2l_b128(&gA[1], &sA[0][sOff + 8]);
  async_g2l_b128(&gB[0], &sB[0][sOff]);
  async_g2l_b128(&gB[1], &sB[0][sOff + 8]);
#endif

  for (int kt = 0; kt < KSTEPS; ++kt) {
    const int cur = kt & 1;
#if USE_ASYNC_LDS
    // issue copy of tile kt+1 into the other buffer (freed by barrier at the
    // end of iteration kt-1), then wait until only those 4 ops are in flight:
    // async loads complete in order, so tile kt's 4 ops are done.
    if (kt + 1 < KSTEPS) {
      const int nxt = 1 - cur;
      async_g2l_b128(&gA[(kt + 1) * 4 + 0], &sA[nxt][sOff]);
      async_g2l_b128(&gA[(kt + 1) * 4 + 1], &sA[nxt][sOff + 8]);
      async_g2l_b128(&gB[(kt + 1) * 4 + 0], &sB[nxt][sOff]);
      async_g2l_b128(&gB[(kt + 1) * 4 + 1], &sB[nxt][sOff + 8]);
      WAIT_ASYNC(4);
    } else {
      WAIT_ASYNC(0);
    }
    __syncthreads();                    // all waves' tile-kt data visible
#else
    const uint4 a0 = gA[kt * 4 + 0];
    const uint4 a1 = gA[kt * 4 + 1];
    const uint4 b0 = gB[kt * 4 + 0];
    const uint4 b1 = gB[kt * 4 + 1];
    if (kt + 1 < KSTEPS) {
      __builtin_prefetch(&gA[kt * 4 + 4], 0, 1);
      __builtin_prefetch(&gB[kt * 4 + 4], 0, 1);
    }
    __syncthreads();
    *(uint4*)&sA[cur][sOff]     = a0;
    *(uint4*)&sA[cur][sOff + 8] = a1;
    *(uint4*)&sB[cur][sOff]     = b0;
    *(uint4*)&sB[cur][sOff + 8] = b1;
    __syncthreads();
#endif

    // --- A fragments (ISA 16-bit A 16x32 layout):
    // lane<16 holds K {kb..kb+7, kb+16..kb+23} with kb = (lane>>4)*8
    v16bf fa[2];
    #pragma unroll
    for (int mi = 0; mi < 2; ++mi) {
      const int rl   = wm * 32 + mi * 16 + lr;
      const int base = rl * 32 + lh * 8;
      union { uint4 u[2]; v16bf v; } tmp;
      tmp.u[0] = *(const uint4*)&sA[cur][base];
      tmp.u[1] = *(const uint4*)&sA[cur][base + 16];
      fa[mi] = tmp.v;
    }
    // --- B fragments: lane holds column n = lr, K {kb..kb+7, kb+16..kb+23} with kb = (lane>>4)*8
    v16bf fb[4];
    #pragma unroll
    for (int ni = 0; ni < 4; ++ni) {
      const int nl   = wn * 64 + ni * 16 + lr;
      const int base = nl * 32 + lh * 8;
      union { uint4 u[2]; v16bf v; } tmp;
      tmp.u[0] = *(const uint4*)&sB[cur][base];
      tmp.u[1] = *(const uint4*)&sB[cur][base + 16];
      fb[ni] = tmp.v;
    }
    #pragma unroll
    for (int mi = 0; mi < 2; ++mi)
      #pragma unroll
      for (int ni = 0; ni < 4; ++ni)
        acc[mi][ni] = __builtin_amdgcn_wmma_f32_16x16x32_bf16(
            false, fa[mi], false, fb[ni], (short)0, acc[mi][ni], false, false);

    __syncthreads();                    // buffer `cur` free for reuse at kt+2
  }

  // --- store (ISA 32-bit C/D 16x16 layout: VGPR r -> M = r + (lane>=16)*8)
  #pragma unroll
  for (int mi = 0; mi < 2; ++mi) {
    #pragma unroll
    for (int ni = 0; ni < 4; ++ni) {
      const int col   = bn * 128 + wn * 64 + ni * 16 + lr;
      const int rbase = bm * 128 + wm * 32 + mi * 16 + lh * 8;
      #pragma unroll
      for (int r = 0; r < 8; ++r)
        C[(size_t)(rbase + r) * NDIM + col] = acc[mi][ni][r];
    }
  }
}

// ---------------- 5) hyperbolic epilogue (per output row) ----------------
__global__ __launch_bounds__(256)
void epilogue_kernel(const float* __restrict__ mx,
                     const float* __restrict__ b,
                     const float* __restrict__ xnorm,
                     const float* __restrict__ b2p,
                     float* __restrict__ out) {
  const int r = blockIdx.x;
  const int t = threadIdx.x;
  const float* row = mx + (size_t)r * NDIM;

  float m[3], bv[3];
  float s1 = 0.f, s2 = 0.f;               // ||mx||^2, <mx,b>
  #pragma unroll
  for (int j = 0; j < 3; ++j) {
    const int d = t + j * 256;
    m[j]  = row[d];
    bv[j] = b[d];
    s1 += m[j] * m[j];
    s2 += m[j] * bv[j];
  }
  s1 = waveReduceSum(s1);
  s2 = waveReduceSum(s2);
  __shared__ float r1[8], r2[8];
  __shared__ float coef[2];
  const int w = t >> 5, l = t & 31;
  if (l == 0) { r1[w] = s1; r2[w] = s2; }
  __syncthreads();

  if (t == 0) {
    float mx2 = 0.f, mxb = 0.f;
    #pragma unroll
    for (int i = 0; i < 8; ++i) { mx2 += r1[i]; mxb += r2[i]; }
    const float b2 = *b2p;

    // ---- mobius_matvec scale (c = 1): res = alpha * mx
    const float xn    = xnorm[r];                       // >= EPSN
    const float mxn_t = sqrtf(mx2);
    const float mxn   = fmaxf(mxn_t, EPSN);
    const float art   = atanhf(fminf(xn, 1.0f - 1e-7f));
    float alpha = tanhf(mxn / xn * art) / mxn;
    if (mx2 == 0.0f) alpha = 0.0f;                      // zero-row rule
    // projx(res)
    const float rn = fmaxf(fabsf(alpha) * mxn_t, EPSN);
    if (rn > MAXNORM) alpha *= MAXNORM / rn;

    // ---- mobius_add(res, b): out_pre = (beta*mx + gamma*b) / denom
    const float x2    = alpha * alpha * mx2;
    const float xy    = alpha * mxb;
    const float beta  = (1.0f + 2.0f * xy + b2) * alpha;
    const float gamma = 1.0f - x2;
    const float denom = fmaxf(1.0f + 2.0f * xy + x2 * b2, EPSN);
    const float bi = beta / denom, gi = gamma / denom;
    // projx(out_pre): ||bi*mx + gi*b||^2 from the same reductions
    const float n2 = bi * bi * mx2 + 2.0f * bi * gi * mxb + gi * gi * b2;
    const float nn = fmaxf(sqrtf(n2), EPSN);
    const float lam = (nn > MAXNORM) ? (MAXNORM / nn) : 1.0f;
    coef[0] = lam * bi;
    coef[1] = lam * gi;
  }
  __syncthreads();

  const float sm = coef[0], sb = coef[1];
  #pragma unroll
  for (int j = 0; j < 3; ++j) {
    const int d = t + j * 256;
    out[(size_t)r * NDIM + d] = sm * m[j] + sb * bv[j];
  }
}

// ---------------------------------------------------------------------------
extern "C" void kernel_launch(void* const* d_in, const int* in_sizes, int n_in,
                              void* d_out, int out_size, void* d_ws, size_t ws_size,
                              hipStream_t stream) {
  const float* x = (const float*)d_in[0];   // [64,3,224,224]
  const float* W = (const float*)d_in[1];   // [768,768]
  const float* b = (const float*)d_in[2];   // [768]
  float* out = (float*)d_out;               // [64,196,768] f32

  // workspace layout (all offsets 16B-aligned); total ~59.04 MB
  char* ws = (char*)d_ws;
  __bf16* patches = (__bf16*)(ws);                       // 12544*768*2 = 19267584
  __bf16* Wbf     = (__bf16*)(ws + 19267584);            //   768*768*2 =  1179648
  float*  mx      = (float*)(ws + 20447232);             // 12544*768*4 = 38535168
  float*  xnorm   = (float*)(ws + 58982400);             // 12544*4     =    50176
  float*  b2      = (float*)(ws + 59032576);             // 4

  patchify_kernel<<<NROWS, 256, 0, stream>>>(x, patches, xnorm);
  convertW_kernel<<<(NDIM * KDIM) / 256, 256, 0, stream>>>(W, Wbf);
  reduceB_kernel<<<1, 256, 0, stream>>>(b, b2);
  gemm_kernel<<<dim3(NDIM / 128, NROWS / 128), 256, 0, stream>>>(patches, Wbf, mx);
  epilogue_kernel<<<NROWS, 256, 0, stream>>>(mx, b, xnorm, b2, out);
}
